// DiagGaussianActor_HR_83726092468897
// MI455X (gfx1250) — matrix operand order repaired and checked by reading, not compiled, not run on hardware
//
#include <hip/hip_runtime.h>
#include <hip/hip_bf16.h>
#include <math.h>
#include <stdint.h>

// ---------------- problem constants ----------------
static constexpr int BB   = 8192;   // batch
static constexpr int OBS  = 256;    // obs dim
static constexpr int ACT  = 32;     // action dim (output = 2*ACT)
static constexpr int HID  = 512;    // hidden
static constexpr int NP   = 8;      // experts
static constexpr float LSMIN = -5.0f;
static constexpr float LSMAX =  2.0f;

// ---------------- WMMA types ----------------
typedef __attribute__((ext_vector_type(16))) __bf16 v16bf;
typedef __attribute__((ext_vector_type(8)))  float  v8f;

union Frag {
    uint4  u[2];
    v16bf  v;
};

// ---------------- TDM availability (device pass only) ----------------
#if defined(__has_builtin)
#  if __has_builtin(__builtin_amdgcn_tensor_load_to_lds) && __has_builtin(__builtin_amdgcn_s_wait_tensorcnt)
#    define USE_TDM 1
#  endif
#endif
#ifndef USE_TDM
#  define USE_TDM 0
#endif

#if USE_TDM
typedef unsigned int u32x4 __attribute__((ext_vector_type(4)));
typedef int          i32x8 __attribute__((ext_vector_type(8)));
typedef int          i32x4 __attribute__((ext_vector_type(4)));

// 2-D TDM tile load: global (row stride = stride0 elems, 2B/elem) -> LDS,
// with LDS padding of (pad_amt_code+1) DWORDs every 2^(pad_int_code+1) DWORDs.
__device__ __forceinline__ void tdm_load_2d(unsigned lds_off, const void* gptr,
                                            unsigned tile_d0, unsigned tile_d1,
                                            unsigned long long stride0,
                                            unsigned tensor_d0, unsigned tensor_d1,
                                            unsigned pad_int_code, unsigned pad_amt_code)
{
    unsigned long long ga = (unsigned long long)(uintptr_t)gptr;
    u32x4 g0;
    g0[0] = 1u;                                   // count=1 (valid user D#)
    g0[1] = lds_off;                              // lds_addr
    g0[2] = (unsigned)(ga & 0xFFFFFFFFu);         // global_addr[31:0]
    g0[3] = (unsigned)((ga >> 32) & 0x1FFFFFFu)   // global_addr[56:32]
          | (2u << 30);                           // type = 2 ("image")
    i32x8 g1;
    g1[0] = (int)((1u << 16)                      // data_size = 2 bytes
                | (1u << 20)                      // pad_enable
                | (pad_int_code << 22)
                | (pad_amt_code << 25));
    g1[1] = (int)((tensor_d0 & 0xFFFFu) << 16);                         // tensor_dim0 lo16
    g1[2] = (int)((tensor_d0 >> 16) | ((tensor_d1 & 0xFFFFu) << 16));   // dim0 hi / dim1 lo
    g1[3] = (int)((tensor_d1 >> 16) | (tile_d0 << 16));                 // dim1 hi / tile_dim0
    g1[4] = (int)(tile_d1 & 0xFFFFu);                                   // tile_dim1 (tile_dim2=0)
    g1[5] = (int)(unsigned)(stride0 & 0xFFFFFFFFull);                   // dim0_stride lo32
    g1[6] = (int)(unsigned)((stride0 >> 32) & 0xFFFFull);               // dim0_stride hi16
    g1[7] = 0;
    i32x4 z4 = {0, 0, 0, 0};
#if __clang_major__ >= 23
    i32x8 z8 = {0, 0, 0, 0, 0, 0, 0, 0};
    __builtin_amdgcn_tensor_load_to_lds(g0, g1, z4, z4, z8, 0);
#else
    __builtin_amdgcn_tensor_load_to_lds(g0, g1, z4, z4, 0);
#endif
}
#endif  // USE_TDM

// ---------------- f32 -> bf16 conversion ----------------
__global__ __launch_bounds__(256) void cvt_f32_bf16(const float* __restrict__ src,
                                                    __bf16* __restrict__ dst, int n) {
    int i = blockIdx.x * 256 + threadIdx.x;
    if (i < n) dst[i] = (__bf16)src[i];
}

// ---------------- fused (blended) WMMA GEMM ----------------
//   acc[b, o] = sum_p blendw[b,p] * ( sum_i W[p,o,i]*A[b,i] + bias[p,o] )
// P==1 => plain GEMM with bias. EPI==0: relu->bf16; EPI==2: actor epilogue->f32.
// A strip (128 x K) is resident in LDS for the whole block; B tiles (64 x 32)
// are double-buffered via the Tensor Data Mover when available.
template<int K, int P, int EPI>
__global__ __launch_bounds__(256)
void moe_wmma_gemm(const __bf16* __restrict__ A,      // [M, K] bf16 row-major
                   const __bf16* __restrict__ W,      // [P, Ntot, K] bf16 (K contiguous)
                   const float*  __restrict__ bias,   // [P, Ntot]
                   const float*  __restrict__ blendw, // [M, 8] (unused if P==1)
                   void* __restrict__ outp,
                   int Ntot)
{
    constexpr int ASTR   = K + 8;          // padded A row stride (elems); 16B pad / row
    constexpr int BS     = 40;             // padded B row stride (elems)
    constexpr int KT     = K / 32;
    constexpr int ABYTES = 128 * ASTR * 2;
    constexpr unsigned A_PAD_CODE = (K == 512) ? 7u : 6u;  // 2^(c+1) DW = K/2 DW

    extern __shared__ __align__(16) char smem[];
    __bf16* Astrip = (__bf16*)smem;
    __bf16* Bbuf0  = (__bf16*)(smem + ABYTES);
    __bf16* Bbuf1  = (__bf16*)(smem + ABYTES + 64 * BS * 2);
    float*  Bl     = (float*) (smem + ABYTES + 2 * 64 * BS * 2);

    const int t       = threadIdx.x;
    const int lane    = t & 31;
    const int wave    = t >> 5;
    const int wm      = wave & 3;   // 32-row strip within 128
    const int wn      = wave >> 2;  // 32-col strip within 64
    const int rowBase = blockIdx.x * 128;
    const int colBase = blockIdx.y * 64;
    const int half    = (lane < 16) ? 0 : 1;
    const int l15     = lane & 15;

    if (P > 1) {
        #pragma unroll
        for (int i = 0; i < 4; ++i) {
            int idx = t + i * 256;
            Bl[idx] = blendw[(size_t)rowBase * 8 + idx];
        }
    }

#if USE_TDM
    if (wave == 0) {
        // resident A strip + first B tile via Tensor Data Mover
        tdm_load_2d((unsigned)(uintptr_t)Astrip, A + (size_t)rowBase * K,
                    (unsigned)K, 128u, (unsigned long long)K,
                    (unsigned)K, (unsigned)BB, A_PAD_CODE, 3u);
        tdm_load_2d((unsigned)(uintptr_t)Bbuf0, W + (size_t)colBase * K,
                    32u, 64u, (unsigned long long)K,
                    (unsigned)K, (unsigned)(P * Ntot), 3u, 3u);
    }
#else
    // cooperative resident A strip load (padded layout)
    for (int ch = t; ch < 128 * K / 8; ch += 256) {
        int r  = ch / (K / 8);
        int kc = ch % (K / 8);
        *(uint4*)&Astrip[r * ASTR + kc * 8] =
            *(const uint4*)&A[(size_t)(rowBase + r) * K + kc * 8];
    }
#endif

    v8f acc[2][2] = {};
    int iter = 0;
    for (int p = 0; p < P; ++p) {
        v8f c[2][2] = {};
        #pragma unroll 1
        for (int kt = 0; kt < KT; ++kt, ++iter) {
            __bf16* Bcur = (iter & 1) ? Bbuf1 : Bbuf0;
#if USE_TDM
            if (wave == 0) __builtin_amdgcn_s_wait_tensorcnt(0);
            __syncthreads();   // Bcur ready for all; all done reading the other buffer
            if (wave == 0 && iter + 1 < P * KT) {
                const int nit = iter + 1;
                const int np  = nit / KT;
                const int nkt = nit % KT;
                __bf16* Bnxt = (nit & 1) ? Bbuf1 : Bbuf0;
                tdm_load_2d((unsigned)(uintptr_t)Bnxt,
                            W + ((size_t)np * Ntot + colBase) * K + nkt * 32,
                            32u, 64u, (unsigned long long)K,
                            (unsigned)K, (unsigned)(P * Ntot), 3u, 3u);
            }
#else
            __syncthreads();
            {
                int br = t >> 2, bkc = t & 3;
                *(uint4*)&Bcur[br * BS + bkc * 8] =
                    *(const uint4*)&W[((size_t)p * Ntot + colBase + br) * K + kt * 32 + bkc * 8];
            }
            __syncthreads();
#endif
            // fragments per ISA layouts:
            // A 16x32 bf16: lanes 0-15 row m, K{0..7,16..23}; lanes 16-31 K{8..15,24..31}
            // B 32x16 bf16: lanes 0-15 col n, K{0..7,16..23}; lanes 16-31 K{8..15,24..31}
            Frag a[2], b[2];
            #pragma unroll
            for (int i = 0; i < 2; ++i) {
                const __bf16* ap = &Astrip[(wm * 32 + i * 16 + l15) * ASTR + kt * 32 + half * 8];
                a[i].u[0] = *(const uint4*)ap;
                a[i].u[1] = *(const uint4*)(ap + 16);
                const __bf16* bp = &Bcur[(wn * 32 + i * 16 + l15) * BS + half * 8];
                b[i].u[0] = *(const uint4*)bp;
                b[i].u[1] = *(const uint4*)(bp + 16);
            }
            #pragma unroll
            for (int i = 0; i < 2; ++i)
                #pragma unroll
                for (int j = 0; j < 2; ++j)
                    c[i][j] = __builtin_amdgcn_wmma_f32_16x16x32_bf16(
                        false, a[i].v, false, b[j].v, (short)0, c[i][j], false, false);
        }

        // fold expert p: acc += blend[b,p] * (c + bias[p, col])
        #pragma unroll
        for (int i = 0; i < 2; ++i) {
            #pragma unroll
            for (int j = 0; j < 2; ++j) {
                const int gcol = colBase + wn * 32 + j * 16 + l15;
                const float bb = bias[(size_t)p * Ntot + gcol];
                #pragma unroll
                for (int r = 0; r < 8; ++r) {
                    float bw = 1.0f;
                    if (P > 1) {
                        int lrow = wm * 32 + i * 16 + r + half * 8;
                        bw = Bl[lrow * 8 + p];
                    }
                    acc[i][j][r] += bw * (c[i][j][r] + bb);
                }
            }
        }
    }

    // epilogue (C layout: VGPR r -> M = base + r (+8 for hi lanes), N = lane&15)
    #pragma unroll
    for (int i = 0; i < 2; ++i) {
        #pragma unroll
        for (int j = 0; j < 2; ++j) {
            const int gcol = colBase + wn * 32 + j * 16 + l15;
            #pragma unroll
            for (int r = 0; r < 8; ++r) {
                const int grow = rowBase + wm * 32 + i * 16 + r + half * 8;
                float v = acc[i][j][r];
                if (EPI == 0) {
                    v = v > 0.0f ? v : 0.0f;
                    ((__bf16*)outp)[(size_t)grow * Ntot + gcol] = (__bf16)v;
                } else {
                    if (gcol >= (Ntot >> 1)) {
                        float tv = tanhf(v);
                        float ls = LSMIN + 0.5f * (LSMAX - LSMIN) * (tv + 1.0f);
                        v = expf(ls);
                    }
                    ((float*)outp)[(size_t)grow * Ntot + gcol] = v;
                }
            }
        }
    }
}

// ---------------- gate head: logits (N=8) + softmax, one wave per row ----------------
__global__ __launch_bounds__(256)
void gate_softmax(const __bf16* __restrict__ h1,   // [BB, HID] bf16
                  const float* __restrict__ gw2,   // [NP, HID]
                  const float* __restrict__ gb2,   // [NP]
                  float* __restrict__ blendw)      // [BB, NP]
{
    const int lane = threadIdx.x & 31;
    const int wave = threadIdx.x >> 5;
    const int row  = blockIdx.x * 8 + wave;

    float s[NP] = {};
    for (int k = lane; k < HID; k += 32) {
        float h = (float)h1[(size_t)row * HID + k];
        #pragma unroll
        for (int p = 0; p < NP; ++p) s[p] += h * gw2[p * HID + k];
    }
    #pragma unroll
    for (int p = 0; p < NP; ++p) {
        #pragma unroll
        for (int off = 16; off > 0; off >>= 1) s[p] += __shfl_xor(s[p], off, 32);
        s[p] += gb2[p];
    }
    float m = s[0];
    #pragma unroll
    for (int p = 1; p < NP; ++p) m = fmaxf(m, s[p]);
    float den = 0.0f;
    #pragma unroll
    for (int p = 0; p < NP; ++p) den += expf(s[p] - m);

    float mine = 0.0f;
    #pragma unroll
    for (int p = 0; p < NP; ++p) mine = (lane == p) ? s[p] : mine;
    if (lane < NP)
        blendw[(size_t)row * NP + lane] = expf(mine - m) / den;
}

// ---------------- host-side launch ----------------
static inline int smem_bytes_for(int K) {
    return 128 * (K + 8) * 2 + 2 * 64 * 40 * 2 + 128 * 8 * 4;
}

extern "C" void kernel_launch(void* const* d_in, const int* in_sizes, int n_in,
                              void* d_out, int out_size, void* d_ws, size_t ws_size,
                              hipStream_t stream) {
    const float* obs = (const float*)d_in[0];
    const float* gw0 = (const float*)d_in[1];
    const float* gb0 = (const float*)d_in[2];
    const float* gw1 = (const float*)d_in[3];
    const float* gb1 = (const float*)d_in[4];
    const float* gw2 = (const float*)d_in[5];
    const float* gb2 = (const float*)d_in[6];
    const float* ew0 = (const float*)d_in[7];
    const float* eb0 = (const float*)d_in[8];
    const float* ew1 = (const float*)d_in[9];
    const float* eb1 = (const float*)d_in[10];
    const float* ew2 = (const float*)d_in[11];
    const float* eb2 = (const float*)d_in[12];

    char*  ws  = (char*)d_ws;
    size_t off = 0;
    auto alloc = [&](size_t bytes) -> char* {
        char* p = ws + off;
        off += (bytes + 255) & ~(size_t)255;
        return p;
    };

    __bf16* obs_bf = (__bf16*)alloc((size_t)BB * OBS * 2);
    __bf16* gw0_bf = (__bf16*)alloc((size_t)HID * OBS * 2);
    __bf16* gw1_bf = (__bf16*)alloc((size_t)HID * HID * 2);
    __bf16* ew0_bf = (__bf16*)alloc((size_t)NP * HID * OBS * 2);
    __bf16* ew1_bf = (__bf16*)alloc((size_t)NP * HID * HID * 2);
    __bf16* ew2_bf = (__bf16*)alloc((size_t)NP * 2 * ACT * HID * 2);
    __bf16* h0     = (__bf16*)alloc((size_t)BB * HID * 2);
    __bf16* h1     = (__bf16*)alloc((size_t)BB * HID * 2);
    __bf16* x1     = (__bf16*)alloc((size_t)BB * HID * 2);
    __bf16* x2     = (__bf16*)alloc((size_t)BB * HID * 2);
    float*  blendw = (float*) alloc((size_t)BB * NP * 4);

    auto cvt = [&](const float* s, __bf16* d, int n) {
        cvt_f32_bf16<<<(n + 255) / 256, 256, 0, stream>>>(s, d, n);
    };
    cvt(obs, obs_bf, BB * OBS);
    cvt(gw0, gw0_bf, HID * OBS);
    cvt(gw1, gw1_bf, HID * HID);
    cvt(ew0, ew0_bf, NP * HID * OBS);
    cvt(ew1, ew1_bf, NP * HID * HID);
    cvt(ew2, ew2_bf, NP * 2 * ACT * HID);

    const int smemK256 = smem_bytes_for(OBS);   // ~80 KB
    const int smemK512 = smem_bytes_for(HID);   // ~144 KB (<= 320 KB/WGP)
    hipFuncSetAttribute((const void*)&moe_wmma_gemm<OBS, 1, 0>,
                        hipFuncAttributeMaxDynamicSharedMemorySize, smemK256);
    hipFuncSetAttribute((const void*)&moe_wmma_gemm<HID, 1, 0>,
                        hipFuncAttributeMaxDynamicSharedMemorySize, smemK512);
    hipFuncSetAttribute((const void*)&moe_wmma_gemm<OBS, NP, 0>,
                        hipFuncAttributeMaxDynamicSharedMemorySize, smemK256);
    hipFuncSetAttribute((const void*)&moe_wmma_gemm<HID, NP, 0>,
                        hipFuncAttributeMaxDynamicSharedMemorySize, smemK512);
    hipFuncSetAttribute((const void*)&moe_wmma_gemm<HID, NP, 2>,
                        hipFuncAttributeMaxDynamicSharedMemorySize, smemK512);

    dim3 blk(256);
    dim3 gFull(BB / 128, HID / 64);   // 64 x 8
    dim3 gOut (BB / 128, 1);          // 64 x 1  (Ntot = 64)

    // gate trunk
    moe_wmma_gemm<OBS, 1, 0><<<gFull, blk, smemK256, stream>>>(obs_bf, gw0_bf, gb0, nullptr, h0, HID);
    moe_wmma_gemm<HID, 1, 0><<<gFull, blk, smemK512, stream>>>(h0,     gw1_bf, gb1, nullptr, h1, HID);
    // gate head + softmax -> blend weights
    gate_softmax<<<BB / 8, blk, 0, stream>>>(h1, gw2, gb2, blendw);
    // blended expert trunk
    moe_wmma_gemm<OBS, NP, 0><<<gFull, blk, smemK256, stream>>>(obs_bf, ew0_bf, eb0, blendw, x1, HID);
    moe_wmma_gemm<HID, NP, 0><<<gFull, blk, smemK512, stream>>>(x1,     ew1_bf, eb1, blendw, x2, HID);
    // final blended layer + actor epilogue (mu | std) -> f32 d_out [BB, 64]
    moe_wmma_gemm<HID, NP, 2><<<gOut,  blk, smemK512, stream>>>(x2,     ew2_bf, eb2, blendw, d_out, 2 * ACT);
}
